// statecomplexrnn_45148696215677
// MI455X (gfx1250) — matrix operand order repaired and checked by reading, not compiled, not run on hardware
//
#include <hip/hip_runtime.h>
#include <stdint.h>

#define TT 8000
#define FF 512
#define UU 512
#define NG 2048   // 4*U gate columns

// ---------- WMMA types (gfx1250, wave32) ----------
typedef __attribute__((ext_vector_type(16))) __bf16 v16bf;
typedef __attribute__((ext_vector_type(8)))  float  v8f;

struct __align__(16) U4 { unsigned x, y, z, w; };
union FragAB { v16bf v; U4 q[2]; };

__device__ __forceinline__ unsigned f32_to_bf16_rne(float f) {
    unsigned u = __float_as_uint(f);
    return (u + 0x7fffu + ((u >> 16) & 1u)) >> 16;
}

// ---------- convert inputs to bf16 (x as-is, W transposed to [N][K]) ----------
__global__ void cvt_x_kernel(const float* __restrict__ xr, const float* __restrict__ xi,
                             unsigned short* __restrict__ xrb, unsigned short* __restrict__ xib,
                             int n) {
    int i = blockIdx.x * blockDim.x + threadIdx.x;
    if (i < n) {
        xrb[i] = (unsigned short)f32_to_bf16_rne(xr[i]);
        xib[i] = (unsigned short)f32_to_bf16_rne(xi[i]);
    }
}

__global__ void cvt_wT_kernel(const float* __restrict__ Wr, const float* __restrict__ Wi,
                              unsigned short* __restrict__ WrT, unsigned short* __restrict__ WiT,
                              int n) {
    int i = blockIdx.x * blockDim.x + threadIdx.x;
    if (i < n) {
        int col = i >> 9;          // 0..2047  (output row of W^T)
        int k   = i & (FF - 1);    // 0..511
        WrT[i] = (unsigned short)f32_to_bf16_rne(Wr[k * NG + col]);
        WiT[i] = (unsigned short)f32_to_bf16_rne(Wi[k * NG + col]);
    }
}

// ---------- phase 1: xz[p] = x_p @ W_p + b_p via v_wmma_f32_16x16x32_bf16 ----------
// Block = 256 threads = 8 waves. Each wave owns one 16x16 output tile; block spans
// 16(M) x 128(N). Grid (500, 16, 4 passes). K loop = 512/32 = 16 WMMAs per wave.
__global__ __launch_bounds__(256) void gemm_xw_kernel(
    const unsigned short* __restrict__ xrb, const unsigned short* __restrict__ xib,
    const unsigned short* __restrict__ WrT, const unsigned short* __restrict__ WiT,
    const float* __restrict__ br, const float* __restrict__ bi,
    float* __restrict__ xz_all)
{
    const int p = blockIdx.z;                 // pass: 0=r2r 1=r2i 2=i2r 3=i2i
    const unsigned short* xb = (p < 2) ? xrb : xib;
    const unsigned short* WT = (p & 1) ? WiT : WrT;
    const float* bias        = (p & 1) ? bi  : br;

    const int m0   = blockIdx.x * 16;
    const int wave = threadIdx.x >> 5;
    const int lane = threadIdx.x & 31;
    const int n0   = blockIdx.y * 128 + wave * 16;
    const int ml   = lane & 15;
    const int hi   = lane >> 4;               // lane half selects K sub-range

    // A fragment: lane row M=ml; low lanes K 0-7/16-23, high lanes K 8-15/24-31
    const unsigned short* arow = xb + (size_t)(m0 + ml) * FF + hi * 8;
    // B fragment (W^T row-major [N][K]): low lanes K 0-7/16-23, high lanes K 8-15/24-31
    const unsigned short* brow = WT + (size_t)(n0 + ml) * FF + hi * 8;

    v8f acc = {};
#pragma unroll
    for (int k0 = 0; k0 < FF; k0 += 32) {
        FragAB a, b;
        a.q[0] = *(const U4*)(arow + k0);
        a.q[1] = *(const U4*)(arow + k0 + 16);
        b.q[0] = *(const U4*)(brow + k0);
        b.q[1] = *(const U4*)(brow + k0 + 16);
        acc = __builtin_amdgcn_wmma_f32_16x16x32_bf16(
            false, a.v, false, b.v, (short)0, acc, false, false);
    }
    // C layout: VGPR j -> row j (lanes 0-15) / j+8 (lanes 16-31), col = lane&15
    const int ncol = n0 + ml;
    const float bv = bias[ncol];
    float* outp = xz_all + (size_t)p * TT * NG + (size_t)(m0 + hi * 8) * NG + ncol;
#pragma unroll
    for (int j = 0; j < 8; ++j)
        outp[(size_t)j * NG] = acc[j] + bv;
}

// ---------- sync-counter init (ws is poisoned 0xAA; re-zero every launch) ----------
__global__ void init_sync_kernel(unsigned* s) {
    if (threadIdx.x < 8) s[threadIdx.x] = 0u;
}

// ---------- phase 2: persistent recurrence ----------
// 32 WGs: pass = blk>>3, wg = blk&7. WG owns 64 units -> 256 gate columns; its
// R slice lives in LDS as packed bf16 k-pairs [256 kpair][256 col] = 256 KB.
// Per step: 256 threads each do one 512-long dot (as 256 packed pairs), gate
// exchange through LDS, h slice published to L2, atomic arrive + spin (8 WGs).
__global__ __launch_bounds__(256) void lstm_scan_kernel(
    const float* __restrict__ Rr, const float* __restrict__ Ri,
    const float* __restrict__ in_states,
    const float* __restrict__ xz_all,
    float* __restrict__ hs_all,
    float* h_buf, unsigned* sync_cnt,
    float* __restrict__ out_states)
{
    extern __shared__ __align__(16) char smem_raw[];
    unsigned* Rp  = (unsigned*)smem_raw;                    // [256][256] u32
    float* h_lds  = (float*)(smem_raw + 256 * 256 * 4);     // 512
    float* z_lds  = h_lds + UU;                             // 256
    float* c_lds  = z_lds + 256;                            // 64

    const int tid = threadIdx.x;
    const int p   = blockIdx.x >> 3;
    const int wg  = blockIdx.x & 7;
    const int u0  = wg * 64;
    const float* R = ((p & 1) == 0) ? Rr : Ri;

    // Load + pack R slice: col c -> gate g=c>>6, unit u0+(c&63); global col g*512+u0+uo
    for (int idx = tid; idx < 256 * 256; idx += 256) {
        int k2 = idx >> 8, c = idx & 255;
        int gcolc = ((c >> 6) << 9) + u0 + (c & 63);
        float a = R[(size_t)(2 * k2)     * NG + gcolc];
        float b = R[(size_t)(2 * k2 + 1) * NG + gcolc];
        Rp[idx] = (f32_to_bf16_rne(b) << 16) | f32_to_bf16_rne(a);
    }
    // h0 / c0 from in_states[0, p, :, {0,1}]
    for (int k = tid; k < UU; k += 256)
        h_lds[k] = in_states[(size_t)(p * UU + k) * 2 + 0];
    if (tid < 64)
        c_lds[tid] = in_states[(size_t)(p * UU + u0 + tid) * 2 + 1];
    __syncthreads();

    const int g    = tid >> 6;
    const int uo   = tid & 63;
    const int gcol = (g << 9) + u0 + uo;
    const float* xz = xz_all + (size_t)p * TT * NG;
    float* hs       = hs_all + (size_t)p * TT * UU;
    unsigned* cnt   = sync_cnt + p;

    for (int t = 0; t < TT; ++t) {
        float acc = xz[(size_t)t * NG + gcol];
        if (t + 1 < TT)  // gfx1250 global_prefetch for next step's activations
            __builtin_prefetch(&xz[(size_t)(t + 1) * NG + gcol], 0, 1);

        const unsigned* rp = Rp + tid;         // lane-consecutive cols: conflict-free
#pragma unroll 8
        for (int k2 = 0; k2 < 256; ++k2) {
            unsigned rv = rp[k2 << 8];
            float2 hv = *(const float2*)&h_lds[k2 << 1];   // broadcast read
            acc += hv.x * __uint_as_float(rv << 16);
            acc += hv.y * __uint_as_float(rv & 0xffff0000u);
        }
        z_lds[tid] = acc;
        __syncthreads();

        if (tid < 64) {
            float zi = z_lds[tid];
            float zf = z_lds[64 + tid];
            float zg = z_lds[128 + tid];
            float zo = z_lds[192 + tid];
            float cp = c_lds[tid];
            float si = 1.f / (1.f + __expf(-zi));
            float sf = 1.f / (1.f + __expf(-zf));
            float so = 1.f / (1.f + __expf(-zo));
            float cn = sf * cp + si * tanhf(zg);
            float hn = so * tanhf(cn);
            c_lds[tid] = cn;
            hs[(size_t)t * UU + u0 + tid] = hn;
            __hip_atomic_store(&h_buf[p * UU + u0 + tid], hn,
                               __ATOMIC_RELAXED, __HIP_MEMORY_SCOPE_AGENT);
        }
        __threadfence();
        __syncthreads();
        if (tid == 0) {
            __hip_atomic_fetch_add(cnt, 1u, __ATOMIC_RELEASE, __HIP_MEMORY_SCOPE_AGENT);
            unsigned tgt = 8u * (unsigned)(t + 1);
            while (__hip_atomic_load(cnt, __ATOMIC_ACQUIRE, __HIP_MEMORY_SCOPE_AGENT) < tgt)
                __builtin_amdgcn_s_sleep(2);
        }
        __syncthreads();
        for (int k = tid; k < UU; k += 256)
            h_lds[k] = __hip_atomic_load(&h_buf[p * UU + k],
                                         __ATOMIC_RELAXED, __HIP_MEMORY_SCOPE_AGENT);
        __syncthreads();
    }

    // out_states[0, p, u, 0]=h  [0, p, u, 1]=c
    if (tid < 64) {
        int u = u0 + tid;
        out_states[(size_t)(p * UU + u) * 2 + 0] = h_lds[u];
        out_states[(size_t)(p * UU + u) * 2 + 1] = c_lds[tid];
    }
}

// ---------- phase 3: real = r2r - i2i ; imag = i2r + r2i ----------
__global__ void combine_kernel(const float* __restrict__ hs_all, float* __restrict__ out) {
    const size_t n = (size_t)TT * UU;
    size_t i = (size_t)blockIdx.x * blockDim.x + threadIdx.x;
    if (i < n) {
        float h0 = hs_all[i];
        float h1 = hs_all[n + i];
        float h2 = hs_all[2 * n + i];
        float h3 = hs_all[3 * n + i];
        out[i]     = h0 - h3;
        out[n + i] = h2 + h1;
    }
}

extern "C" void kernel_launch(void* const* d_in, const int* in_sizes, int n_in,
                              void* d_out, int out_size, void* d_ws, size_t ws_size,
                              hipStream_t stream) {
    const float* in_real   = (const float*)d_in[0];
    const float* in_imag   = (const float*)d_in[1];
    const float* in_states = (const float*)d_in[2];
    const float* Wr = (const float*)d_in[3];
    const float* Rr = (const float*)d_in[4];
    const float* br = (const float*)d_in[5];
    const float* Wi = (const float*)d_in[6];
    const float* Ri = (const float*)d_in[7];
    const float* bi = (const float*)d_in[8];
    float* out = (float*)d_out;

    char* w = (char*)d_ws;
    size_t off = 0;
    auto carve = [&](size_t bytes) -> char* {
        char* pp = w + off;
        off = (off + bytes + 255) & ~(size_t)255;
        return pp;
    };
    unsigned short* xrb = (unsigned short*)carve((size_t)TT * FF * 2);
    unsigned short* xib = (unsigned short*)carve((size_t)TT * FF * 2);
    unsigned short* WrT = (unsigned short*)carve((size_t)NG * FF * 2);
    unsigned short* WiT = (unsigned short*)carve((size_t)NG * FF * 2);
    float* xz_all = (float*)carve((size_t)4 * TT * NG * 4);
    float* hs_all = (float*)carve((size_t)4 * TT * UU * 4);
    float* h_buf  = (float*)carve((size_t)4 * UU * 4);
    unsigned* sync_cnt = (unsigned*)carve(256);

    init_sync_kernel<<<1, 64, 0, stream>>>(sync_cnt);

    int nx = TT * FF;
    cvt_x_kernel<<<(nx + 255) / 256, 256, 0, stream>>>(in_real, in_imag, xrb, xib, nx);
    int nw = NG * FF;
    cvt_wT_kernel<<<(nw + 255) / 256, 256, 0, stream>>>(Wr, Wi, WrT, WiT, nw);

    gemm_xw_kernel<<<dim3(TT / 16, NG / 128, 4), 256, 0, stream>>>(
        xrb, xib, WrT, WiT, br, bi, xz_all);

    size_t smem = (size_t)256 * 256 * 4 + (UU + 256 + 64) * 4;  // ~259.25 KB < 320 KB/WGP
    hipFuncSetAttribute((const void*)lstm_scan_kernel,
                        hipFuncAttributeMaxDynamicSharedMemorySize, (int)smem);
    lstm_scan_kernel<<<32, 256, smem, stream>>>(
        Rr, Ri, in_states, xz_all, hs_all, h_buf, sync_cnt,
        out + 2 * (size_t)TT * UU);

    combine_kernel<<<((TT * UU) + 255) / 256, 256, 0, stream>>>(hs_all, out);
}
